// LearnableCorrBlock_61907658604740
// MI455X (gfx1250) — matrix operand order repaired and checked by reading, not compiled, not run on hardware
//
#include <hip/hip_runtime.h>
#include <hip/hip_bf16.h>
#include <math.h>

// Problem constants (from reference)
#define BB 4
#define CC 256
#define HH 96
#define WW 128
#define KD 9          // 2*4+1 window
#define PLANES 81     // 9*9

typedef __attribute__((ext_vector_type(16))) _Float16 v16h;
typedef __attribute__((ext_vector_type(8)))  _Float16 v8h;
typedef __attribute__((ext_vector_type(8)))  float    v8f;

#define SHUF16(lo, hi) __builtin_shufflevector((lo), (hi), 0,1,2,3,4,5,6,7,8,9,10,11,12,13,14,15)

// ---------------------------------------------------------------------------
// K1: build I-S (augmented with I) and I+S, then Gauss-Jordan invert I-S.
// Single workgroup, 1024 threads.  I-S has eigenvalues 1 - i*lambda (skew S),
// |.| >= 1, so unpivoted elimination is numerically safe.
// ---------------------------------------------------------------------------
__global__ void __launch_bounds__(1024)
k_metric_invert(const float* __restrict__ rp, float* __restrict__ Aaug,
                float* __restrict__ M2) {
    const int tid = threadIdx.x;
    for (int idx = tid; idx < 256 * 512; idx += 1024) {
        const int i = idx >> 9, j = idx & 511;
        if (j < 256) {
            float s;
            if (i == j)      s = 0.0f;
            else if (i < j)  s =  0.5f * rp[i * 256 + j];
            else             s = -0.5f * rp[j * 256 + i];
            const float e = (i == j) ? 1.0f : 0.0f;
            Aaug[i * 512 + j] = e - s;   // I - S
            M2[i * 256 + j]   = e + s;   // I + S
        } else {
            Aaug[i * 512 + j] = ((j - 256) == i) ? 1.0f : 0.0f; // augmented I
        }
    }
    __syncthreads();

    __shared__ float colf[256];
    __shared__ float invp;
    for (int p = 0; p < 256; ++p) {
        if (tid == 0) invp = 1.0f / Aaug[p * 512 + p];
        __syncthreads();
        const float ip = invp;
        for (int j = tid; j < 512; j += 1024) Aaug[p * 512 + j] *= ip;
        __syncthreads();
        if (tid < 256) colf[tid] = Aaug[tid * 512 + p];
        __syncthreads();
        for (int idx = tid; idx < 256 * 512; idx += 1024) {
            const int i = idx >> 9, j = idx & 511;
            if (i != p) Aaug[i * 512 + j] -= colf[i] * Aaug[p * 512 + j];
        }
        __syncthreads();
    }
}

// ---------------------------------------------------------------------------
// K2: P = (I+S) @ inv(I-S).   block = row i, thread = col j.
// ---------------------------------------------------------------------------
__global__ void __launch_bounds__(256)
k_cayley_P(const float* __restrict__ M2, const float* __restrict__ Aaug,
           float* __restrict__ P) {
    const int i = blockIdx.x, j = threadIdx.x;
    __shared__ float m2row[256];
    m2row[j] = M2[i * 256 + j];
    __syncthreads();
    float acc = 0.0f;
    for (int k = 0; k < 256; ++k)
        acc += m2row[k] * Aaug[k * 512 + 256 + j];
    P[i * 256 + j] = acc;
}

// ---------------------------------------------------------------------------
// K3: W16T[d][c] = (f16) sum_k P[k,c] * D[k] * P[k,d]   (= Wm[c][d])
// Stored transposed + channel-contiguous so it serves directly as the
// B-matrix of the projection GEMM (contiguous K per lane).
// ---------------------------------------------------------------------------
__global__ void __launch_bounds__(256)
k_metric_W16T(const float* __restrict__ P, const float* __restrict__ rawD,
              _Float16* __restrict__ W16T) {
    const int d = blockIdx.x, c = threadIdx.x;
    __shared__ float Dv[256];
    {
        const float tD = atanf(rawD[c]) * 0.63661977236758134308f; // 2/pi
        Dv[c] = (1.0f + tD) / (1.0f - tD);
    }
    __syncthreads();
    float acc = 0.0f;
    for (int k = 0; k < 256; ++k)
        acc += P[k * 256 + c] * Dv[k] * P[k * 256 + d];
    W16T[(size_t)d * 256 + c] = (_Float16)acc;
}

// ---------------------------------------------------------------------------
// K4: NCHW f32 -> NHWC f16 for fmap1 and fmap2.
// block=(y,b), 256 threads = channel. Per-thread x-streaming keeps reads
// line-local; writes are fully coalesced (512B per pixel across threads).
// ---------------------------------------------------------------------------
__global__ void __launch_bounds__(256)
k_to_nhwc_f16(const float* __restrict__ f1, const float* __restrict__ f2,
              _Float16* __restrict__ f1h, _Float16* __restrict__ f2h) {
    const int y = blockIdx.x, b = blockIdx.y, c = threadIdx.x;
    const float* s1 = f1 + (((size_t)b * CC + c) * HH + y) * WW;
    const float* s2 = f2 + (((size_t)b * CC + c) * HH + y) * WW;
    _Float16* d1 = f1h + ((size_t)(b * HH + y) * WW) * CC + c;
    _Float16* d2 = f2h + ((size_t)(b * HH + y) * WW) * CC + c;
    for (int x = 0; x < WW; ++x) {
        d1[(size_t)x * CC] = (_Float16)s1[x];
        d2[(size_t)x * CC] = (_Float16)s2[x];
    }
}

// ---------------------------------------------------------------------------
// K5: projection GEMM via WMMA:  proj16[b,y,x,d] = sum_c f1h[b,y,x,c]*Wm[c,d]
// One wave per (b, y, 16-pixel x-tile). A fragments (16x256 f16) hoisted
// into registers; loop 16 N-tiles x 8 K-steps of v_wmma_f32_16x16x32_f16.
// ---------------------------------------------------------------------------
__global__ void __launch_bounds__(32)
k_proj_wmma(const _Float16* __restrict__ f1h, const _Float16* __restrict__ W16T,
            _Float16* __restrict__ ph) {
    const int x0 = blockIdx.x * 16, y = blockIdx.y, b = blockIdx.z;
    const int lane = threadIdx.x;
    const int m = lane & 15, hh = lane >> 4;

    const _Float16* pA = f1h + ((size_t)(b * HH + y) * WW + x0) * CC;
    _Float16*       pD = ph  + ((size_t)(b * HH + y) * WW + x0) * CC;

    // A-matrix fragments: lanes 0-15 rows M with K 0-15; lanes 16-31 K 16-31.
    v16h afr[8];
#pragma unroll
    for (int ks = 0; ks < 8; ++ks) {
        const _Float16* a0 = pA + (size_t)m * CC + ks * 32 + 16 * hh;
        v8h lo = *(const v8h*)a0;
        v8h hi = *(const v8h*)(a0 + 8);
        afr[ks] = SHUF16(lo, hi);
    }

    __shared__ alignas(32) _Float16 st[16][16];

    for (int nt = 0; nt < 16; ++nt) {
        const int n0 = nt * 16;
        v8f acc = {};
#pragma unroll
        for (int ks = 0; ks < 8; ++ks) {
            // B-matrix: lanes 0-15 hold K 0-15 of column n, lanes 16-31 K 16-31.
            const _Float16* bp = W16T + (size_t)(n0 + m) * CC + ks * 32 + 16 * hh;
            v8h blo = *(const v8h*)bp;
            v8h bhi = *(const v8h*)(bp + 8);
            v16h bf = SHUF16(blo, bhi);
            acc = __builtin_amdgcn_wmma_f32_16x16x32_f16(
                false, afr[ks], false, bf, (short)0, acc, false, false);
        }
        // De-swizzle D tile through LDS, emit 16B/lane coalesced f16 stores.
#pragma unroll
        for (int r = 0; r < 8; ++r) st[r + 8 * hh][m] = (_Float16)acc[r];
        __syncthreads();
        v8h ov = *(const v8h*)&st[m][8 * hh];
        *(v8h*)(pD + (size_t)m * CC + n0 + 8 * hh) = ov;
        __syncthreads();
    }
}

// ---------------------------------------------------------------------------
// K6: banded correlation via WMMA.
// One wave per (b, y, x-tile, dy). Band |dx|<=4 of the per-row cross
// correlation is covered by two 16x16 N-tiles at x-offsets -8 and +8.
// corr[b, dy*9+dx, y, x0+m] = X[m, m+dx] / 256.
// ---------------------------------------------------------------------------
__global__ void __launch_bounds__(32)
k_corr_wmma(const _Float16* __restrict__ ph, const _Float16* __restrict__ f2h,
            float* __restrict__ out) {
    const int x0 = blockIdx.x * 16, y = blockIdx.y, z = blockIdx.z;
    const int b = z / KD, dyi = z % KD;
    const int y2 = y + dyi - 4;
    const int lane = threadIdx.x;
    const int m = lane & 15, hh = lane >> 4;

    if (y2 < 0 || y2 >= HH) {   // vertical zero padding: whole dy slab is 0
#pragma unroll
        for (int j = 0; j < 5; ++j) {
            const int idx = j * 32 + lane;
            const int dxi = idx >> 4, mm = idx & 15;
            if (dxi < KD)
                out[(((size_t)b * PLANES + dyi * KD + dxi) * HH + y) * WW + x0 + mm] = 0.0f;
        }
        return;
    }

    const _Float16* pA = ph + ((size_t)(b * HH + y) * WW + x0) * CC;
    v16h afr[8];
#pragma unroll
    for (int ks = 0; ks < 8; ++ks) {
        const _Float16* a0 = pA + (size_t)m * CC + ks * 32 + 16 * hh;
        v8h lo = *(const v8h*)a0;
        v8h hi = *(const v8h*)(a0 + 8);
        afr[ks] = SHUF16(lo, hi);
    }

    const _Float16* pB = f2h + ((size_t)(b * HH + y2) * WW) * CC;
    const int xa = x0 - 8 + m;                 // N-tile a column -> f2 x
    const int xb = x0 + 8 + m;                 // N-tile b column -> f2 x
    const bool va = (xa >= 0) && (xa < WW);    // horizontal zero padding
    const bool vb = (xb < WW);

    v8f acca = {}, accb = {};
#pragma unroll
    for (int ks = 0; ks < 8; ++ks) {
        const int k0 = ks * 32 + 16 * hh;
        v16h bfa = {};
        v16h bfb = {};
        if (va) {
            const _Float16* p0 = pB + (size_t)xa * CC + k0;
            bfa = SHUF16(*(const v8h*)p0, *(const v8h*)(p0 + 8));
        }
        if (vb) {
            const _Float16* p1 = pB + (size_t)xb * CC + k0;
            bfb = SHUF16(*(const v8h*)p1, *(const v8h*)(p1 + 8));
        }
        acca = __builtin_amdgcn_wmma_f32_16x16x32_f16(
            false, afr[ks], false, bfa, (short)0, acca, false, false);
        accb = __builtin_amdgcn_wmma_f32_16x16x32_f16(
            false, afr[ks], false, bfb, (short)0, accb, false, false);
    }

    __shared__ float tile[2][16][16];
#pragma unroll
    for (int r = 0; r < 8; ++r) {
        tile[0][r + 8 * hh][m] = acca[r];
        tile[1][r + 8 * hh][m] = accb[r];
    }
    __syncthreads();

    const float sc = 1.0f / 256.0f;
#pragma unroll
    for (int j = 0; j < 5; ++j) {
        const int idx = j * 32 + lane;
        const int dxi = idx >> 4, mm = idx & 15;
        if (dxi < KD) {
            const int u = mm + dxi + 4;        // = (m + dxoff) + 8, in [4,27]
            const float v = tile[u >> 4][mm][u & 15] * sc;
            out[(((size_t)b * PLANES + dyi * KD + dxi) * HH + y) * WW + x0 + mm] = v;
        }
    }
}

// ---------------------------------------------------------------------------
// Launch
// ---------------------------------------------------------------------------
extern "C" void kernel_launch(void* const* d_in, const int* in_sizes, int n_in,
                              void* d_out, int out_size, void* d_ws, size_t ws_size,
                              hipStream_t stream) {
    const float* f1 = (const float*)d_in[0];
    const float* f2 = (const float*)d_in[1];
    const float* rp = (const float*)d_in[2];
    const float* rd = (const float*)d_in[3];
    float* out = (float*)d_out;

    char* ws = (char*)d_ws;
    float*    Aaug = (float*)(ws);                        //  524288 B (256x512 f32)
    float*    M2   = (float*)(ws + 524288);               //  262144 B
    float*    P    = (float*)(ws + 786432);               //  262144 B
    _Float16* W16T = (_Float16*)(ws + 1048576);           //  131072 B
    _Float16* f1h  = (_Float16*)(ws + 1179648);           // 25165824 B (NHWC f16)
    _Float16* f2h  = (_Float16*)(ws + 26345472);          // 25165824 B
    _Float16* ph   = (_Float16*)(ws + 51511296);          // 25165824 B -> 76.7 MB total

    // metric chain (tiny) — runs concurrently-in-order with the converts
    k_metric_invert<<<1, 1024, 0, stream>>>(rp, Aaug, M2);
    k_cayley_P<<<256, 256, 0, stream>>>(M2, Aaug, P);
    k_metric_W16T<<<256, 256, 0, stream>>>(P, rd, W16T);

    // f32 NCHW -> f16 NHWC
    k_to_nhwc_f16<<<dim3(HH, BB), 256, 0, stream>>>(f1, f2, f1h, f2h);

    // projection GEMM (WMMA)
    k_proj_wmma<<<dim3(WW / 16, HH, BB), 32, 0, stream>>>(f1h, W16T, ph);

    // banded 9x9 correlation (WMMA)
    k_corr_wmma<<<dim3(WW / 16, HH, BB * KD), 32, 0, stream>>>(ph, f2h, out);
}
